// SVD_52415780880777
// MI455X (gfx1250) — matrix operand order repaired and checked by reading, not compiled, not run on hardware
//
#include <hip/hip_runtime.h>
#include <hip/hip_bf16.h>
#include <stdint.h>

// Shapes fixed by the reference: B=16, N=2048, F=512
#define BB 16
#define NN 2048
#define FF 512

typedef __attribute__((ext_vector_type(16))) __bf16 v16bf;
typedef __attribute__((ext_vector_type(8)))  __bf16 v8bf;
typedef __attribute__((ext_vector_type(8)))  float  v8f;
typedef int vsi4 __attribute__((vector_size(16)));   // matches builtin's int4

#if __has_builtin(__builtin_amdgcn_global_load_async_to_lds_b128)
#define HAVE_ASYNC_LDS 1
#endif

// Typed address-space cast helpers (AMDGPU generic LDS pointers carry the LDS
// byte offset in the low 32 bits; generic global pointers are the 64-bit VA).
#define GPTR128(p) ((__attribute__((address_space(1))) vsi4*)(uintptr_t)(p))
#define GPTR32(p)  ((__attribute__((address_space(1))) int*)(uintptr_t)(p))
#define LPTR128(p) ((__attribute__((address_space(3))) vsi4*)(uint32_t)(uintptr_t)(p))
#define LPTR32(p)  ((__attribute__((address_space(3))) int*)(uint32_t)(uintptr_t)(p))

#ifdef HAVE_ASYNC_LDS
__device__ __forceinline__ void wait_async0() {
#if __has_builtin(__builtin_amdgcn_s_wait_asynccnt)
    __builtin_amdgcn_s_wait_asynccnt(0);
#else
    asm volatile("s_wait_asynccnt 0" ::: "memory");
#endif
}
#else
__device__ __forceinline__ void wait_async0() {}
#endif

__device__ __forceinline__ uint16_t f2bf(float f) {
    uint32_t u = __float_as_uint(f);
    uint32_t r = u + 0x7FFFu + ((u >> 16) & 1u);  // round-to-nearest-even
    return (uint16_t)(r >> 16);
}

// ---------------- prep kernels ----------------

__global__ void zero_kernel(float* g) { g[threadIdx.x] = 0.f; }

// fsource (B,F,N) f32 -> bf16, same layout (B-fragment friendly: for a fixed
// feature row k, 16 source columns are contiguous).
__global__ void cvt_kernel(const float* __restrict__ in, uint16_t* __restrict__ out, int n4) {
    int i = blockIdx.x * blockDim.x + threadIdx.x;
    if (i < n4) {
        float4 v = ((const float4*)in)[i];
        uint32_t lo = (uint32_t)f2bf(v.x) | ((uint32_t)f2bf(v.y) << 16);
        uint32_t hi = (uint32_t)f2bf(v.z) | ((uint32_t)f2bf(v.w) << 16);
        ((uint2*)out)[i] = make_uint2(lo, hi);
    }
}

// ftarget (B,F,N) f32 -> (B,N,F) bf16 (A-fragment friendly). 32x32 LDS tile.
__global__ void tr_kernel(const float* __restrict__ in, uint16_t* __restrict__ out) {
    __shared__ float tile[32][33];
    const int b  = blockIdx.z;
    const int i0 = blockIdx.x * 32;   // point (N) tile
    const int f0 = blockIdx.y * 32;   // feature (F) tile
    const int tx = threadIdx.x, ty = threadIdx.y;   // 32 x 8
    const float* src = in + (size_t)b * FF * NN;
#pragma unroll
    for (int r = 0; r < 4; ++r) {
        int f = f0 + ty + r * 8;
        tile[ty + r * 8][tx] = src[(size_t)f * NN + i0 + tx];
    }
    __syncthreads();
    uint16_t* dst = out + (size_t)b * NN * FF;
#pragma unroll
    for (int r = 0; r < 4; ++r) {
        int i = i0 + ty + r * 8;
        dst[(size_t)i * FF + f0 + tx] = f2bf(tile[tx][ty + r * 8]);
    }
}

// target (B,N,3) -> tgtT (B,3,N): per-tile coordinate slices become contiguous.
__global__ void tgt_kernel(const float* __restrict__ target, float* __restrict__ tgtT) {
    int i = blockIdx.x * 256 + threadIdx.x;   // over B*N
    if (i < BB * NN) {
        int b = i / NN, n = i % NN;
#pragma unroll
        for (int c = 0; c < 3; ++c)
            tgtT[((size_t)b * 3 + c) * NN + n] = target[(size_t)i * 3 + c];
    }
}

// ---------------- fused logits + softmax + weighted-sum kernel ----------------
//
// grid: (NN/128, BB); block: 256 threads = 8 waves. Each wave owns 16 source
// columns, B panel (16 cols x K=512 bf16) register-resident in 16 fragments.
// Streams 128 target tiles; tile it+1 is DMA'd (async-to-LDS, ASYNCcnt) into
// the alternate LDS buffer while tile it runs its 16 WMMAs + online softmax.
__global__ __launch_bounds__(256) void reg_kernel(
        const uint16_t* __restrict__ Abf,   // (B,N,F) bf16  = ftarget^T rows
        const uint16_t* __restrict__ Bbf,   // (B,F,N) bf16  = fsource
        const float*    __restrict__ tgtT,  // (B,3,N)
        const float*    __restrict__ source,// (B,N,3)
        float*          __restrict__ gacc)  // (B,16) accumulators
{
    __shared__ __align__(16) uint16_t Atile[2][16 * FF];  // 2 x 16 KB
    __shared__ __align__(16) float sTgt[2][64];
    __shared__ float accS[16];

    const int b     = blockIdx.y;
    const int wave  = threadIdx.x >> 5;
    const int lane  = threadIdx.x & 31;
    const int col   = lane & 15;   // column within wave's 16-col slab
    const int hb    = lane >> 4;   // 0: rows 0-7 of tile, 1: rows 8-15
    const int jwave = blockIdx.x * 128 + wave * 16;

    if (threadIdx.x < 16) accS[threadIdx.x] = 0.f;

    // Stage one 16-row target tile (16 KB of features + 48 tgt floats).
    auto stage = [&](int buf, int ibase) {
        const char* asrc = (const char*)(Abf + ((size_t)b * NN + ibase) * FF);
#ifdef HAVE_ASYNC_LDS
        char* adst = (char*)Atile[buf];
#pragma unroll
        for (int t = 0; t < 4; ++t) {
            int off = threadIdx.x * 16 + t * 4096;
            __builtin_amdgcn_global_load_async_to_lds_b128(
                GPTR128(asrc + off), LPTR128(adst + off), 0, 0);
        }
        if (threadIdx.x < 48) {
            int c = threadIdx.x >> 4, r = threadIdx.x & 15;
            const float* g = tgtT + ((size_t)b * 3 + c) * NN + ibase + r;
#if __has_builtin(__builtin_amdgcn_global_load_async_to_lds_b32)
            __builtin_amdgcn_global_load_async_to_lds_b32(
                GPTR32(g), LPTR32(&sTgt[buf][c * 16 + r]), 0, 0);
#else
            sTgt[buf][c * 16 + r] = *g;
#endif
        }
#else
        const uint4* a4 = (const uint4*)asrc;
        uint4* d4 = (uint4*)Atile[buf];
#pragma unroll
        for (int t = 0; t < 4; ++t)
            d4[threadIdx.x + t * 256] = a4[threadIdx.x + t * 256];
        if (threadIdx.x < 48) {
            int c = threadIdx.x >> 4, r = threadIdx.x & 15;
            sTgt[buf][c * 16 + r] = tgtT[((size_t)b * 3 + c) * NN + ibase + r];
        }
#endif
    };

    // Preload this wave's full B panel into 16 register fragments.
    // B-frag layout: lane%16 = column, 16 K rows per lane in table order.
    v16bf Bfr[16];
#pragma unroll
    for (int kk = 0; kk < 16; ++kk) {
        const __bf16* bp = (const __bf16*)Bbf +
            ((size_t)b * FF + kk * 32 + hb * 8) * NN + jwave + col;
#pragma unroll
        for (int e = 0; e < 16; ++e)
            Bfr[kk][e] = bp[(size_t)((e & 7) + (e >> 3) * 16) * NN];
    }

    stage(0, 0);
    wait_async0();
    __syncthreads();

    float m = -3.0e38f, s = 0.f, p0 = 0.f, p1 = 0.f, p2 = 0.f;

    for (int it = 0; it < NN / 16; ++it) {
        const int cur = it & 1;
        if (it + 1 < NN / 16) stage(1 - cur, (it + 1) * 16);

        // 16x16 logit block, fp32 accumulate over K=512; A-frag double buffer.
        const __bf16* arow = (const __bf16*)Atile[cur] + col * FF + hb * 8;
        v8bf alo = *(const v8bf*)(arow);
        v8bf ahi = *(const v8bf*)(arow + 16);
        v8f acc = {};
#pragma unroll
        for (int kk = 0; kk < 16; ++kk) {
            v16bf af = __builtin_shufflevector(alo, ahi,
                0, 1, 2, 3, 4, 5, 6, 7, 8, 9, 10, 11, 12, 13, 14, 15);
            if (kk < 15) {  // prefetch next fragment before the dependent WMMA
                alo = *(const v8bf*)(arow + (kk + 1) * 32);
                ahi = *(const v8bf*)(arow + (kk + 1) * 32 + 16);
            }
            acc = __builtin_amdgcn_wmma_f32_16x16x32_bf16(
                false, af, false, Bfr[kk], (short)0, acc, false, false);
        }

        // Online softmax over the 8 rows this lane holds of its column.
        const float* tg = sTgt[cur];
        float tmax = acc[0];
#pragma unroll
        for (int v = 1; v < 8; ++v) tmax = fmaxf(tmax, acc[v]);
        float newm = fmaxf(m, tmax);
        float sc = __expf(m - newm);
        s *= sc; p0 *= sc; p1 *= sc; p2 *= sc;
#pragma unroll
        for (int v = 0; v < 8; ++v) {
            int r = v + hb * 8;
            float e = __expf(acc[v] - newm);
            s  += e;
            p0 += e * tg[r];
            p1 += e * tg[16 + r];
            p2 += e * tg[32 + r];
        }
        m = newm;

        wait_async0();     // tile it+1 DMA done (overlapped with the WMMAs above)
        __syncthreads();   // all waves done reading buf[cur], writes visible
    }

    // Merge half-lane partials (rows 0-7 with rows 8-15 of the same column)
    float om = __shfl_xor(m,  16, 32);
    float os = __shfl_xor(s,  16, 32);
    float o0 = __shfl_xor(p0, 16, 32);
    float o1 = __shfl_xor(p1, 16, 32);
    float o2 = __shfl_xor(p2, 16, 32);
    float mm = fmaxf(m, om);
    float wa = __expf(m - mm), wb = __expf(om - mm);
    s  = s * wa  + os * wb;
    p0 = p0 * wa + o0 * wb;
    p1 = p1 * wa + o1 * wb;
    p2 = p2 * wa + o2 * wb;

    if (hb == 0) {
        float inv = 1.f / s;
        float q0 = p0 * inv, q1 = p1 * inv, q2 = p2 * inv;  // pred_tgt[:, j]
        int j = jwave + col;
        float s0 = source[((size_t)b * NN + j) * 3 + 0];
        float s1 = source[((size_t)b * NN + j) * 3 + 1];
        float s2 = source[((size_t)b * NN + j) * 3 + 2];
        atomicAdd(&accS[0], q0);  atomicAdd(&accS[1], q1);  atomicAdd(&accS[2], q2);
        atomicAdd(&accS[3], s0);  atomicAdd(&accS[4], s1);  atomicAdd(&accS[5], s2);
        atomicAdd(&accS[6],  s0 * q0); atomicAdd(&accS[7],  s0 * q1); atomicAdd(&accS[8],  s0 * q2);
        atomicAdd(&accS[9],  s1 * q0); atomicAdd(&accS[10], s1 * q1); atomicAdd(&accS[11], s1 * q2);
        atomicAdd(&accS[12], s2 * q0); atomicAdd(&accS[13], s2 * q1); atomicAdd(&accS[14], s2 * q2);
    }
    __syncthreads();
    if (threadIdx.x < 15) atomicAdd(&gacc[b * 16 + threadIdx.x], accS[threadIdx.x]);
}

// ---------------- per-batch 3x3 SVD / Kabsch solve ----------------

__device__ __forceinline__ float det3(const float M[3][3]) {
    return M[0][0] * (M[1][1] * M[2][2] - M[1][2] * M[2][1])
         - M[0][1] * (M[1][0] * M[2][2] - M[1][2] * M[2][0])
         + M[0][2] * (M[1][0] * M[2][1] - M[1][1] * M[2][0]);
}

__global__ void solve_kernel(const float* __restrict__ gacc, float* __restrict__ out) {
    int b = threadIdx.x;
    if (b >= BB) return;
    const float* g = gacc + b * 16;
    const float invN = 1.f / (float)NN;
    float pmv[3] = { g[0] * invN, g[1] * invN, g[2] * invN };  // pred_tgt mean
    float smv[3] = { g[3] * invN, g[4] * invN, g[5] * invN };  // src mean

    float S[3][3];
#pragma unroll
    for (int c = 0; c < 3; ++c)
#pragma unroll
        for (int d = 0; d < 3; ++d)
            S[c][d] = g[6 + c * 3 + d] - (float)NN * smv[c] * pmv[d];

    float A[3][3];
#pragma unroll
    for (int i = 0; i < 3; ++i)
#pragma unroll
        for (int j = 0; j < 3; ++j)
            A[i][j] = S[0][i] * S[0][j] + S[1][i] * S[1][j] + S[2][i] * S[2][j];

    float V[3][3] = { {1.f, 0.f, 0.f}, {0.f, 1.f, 0.f}, {0.f, 0.f, 1.f} };
    const int PP[3] = {0, 0, 1}, QQ[3] = {1, 2, 2};
    for (int sweep = 0; sweep < 10; ++sweep) {
#pragma unroll
        for (int t = 0; t < 3; ++t) {
            int p = PP[t], q = QQ[t];
            float apq = A[p][q];
            if (fabsf(apq) < 1e-30f) continue;
            float theta = 0.5f * (A[q][q] - A[p][p]) / apq;
            float tt = ((theta >= 0.f) ? 1.f : -1.f) /
                       (fabsf(theta) + sqrtf(theta * theta + 1.f));
            float cc = 1.f / sqrtf(tt * tt + 1.f);
            float ss = tt * cc;
#pragma unroll
            for (int k = 0; k < 3; ++k) {
                float akp = A[k][p], akq = A[k][q];
                A[k][p] = cc * akp - ss * akq;
                A[k][q] = ss * akp + cc * akq;
            }
#pragma unroll
            for (int k = 0; k < 3; ++k) {
                float apk = A[p][k], aqk = A[q][k];
                A[p][k] = cc * apk - ss * aqk;
                A[q][k] = ss * apk + cc * aqk;
            }
#pragma unroll
            for (int k = 0; k < 3; ++k) {
                float vkp = V[k][p], vkq = V[k][q];
                V[k][p] = cc * vkp - ss * vkq;
                V[k][q] = ss * vkp + cc * vkq;
            }
        }
    }

    float ev[3] = { A[0][0], A[1][1], A[2][2] };
#pragma unroll
    for (int a = 0; a < 2; ++a)
#pragma unroll
        for (int c2 = a + 1; c2 < 3; ++c2)
            if (ev[c2] > ev[a]) {
                float te = ev[a]; ev[a] = ev[c2]; ev[c2] = te;
#pragma unroll
                for (int k = 0; k < 3; ++k) {
                    float tv = V[k][a]; V[k][a] = V[k][c2]; V[k][c2] = tv;
                }
            }

    float U[3][3];
#pragma unroll
    for (int k = 0; k < 3; ++k) {
        float sig = sqrtf(fmaxf(ev[k], 0.f));
        float inv = 1.f / fmaxf(sig, 1e-20f);
#pragma unroll
        for (int r = 0; r < 3; ++r)
            U[r][k] = (S[r][0] * V[0][k] + S[r][1] * V[1][k] + S[r][2] * V[2][k]) * inv;
    }

    float d = det3(U) * det3(V);  // det(V U^T)
    float R[3][3];
#pragma unroll
    for (int i = 0; i < 3; ++i)
#pragma unroll
        for (int j = 0; j < 3; ++j)
            R[i][j] = V[i][0] * U[j][0] + V[i][1] * U[j][1] + d * V[i][2] * U[j][2];

#pragma unroll
    for (int i = 0; i < 3; ++i)
#pragma unroll
        for (int j = 0; j < 3; ++j)
            out[b * 9 + i * 3 + j] = R[i][j];
#pragma unroll
    for (int i = 0; i < 3; ++i)
        out[BB * 9 + b * 3 + i] =
            pmv[i] - (R[i][0] * smv[0] + R[i][1] * smv[1] + R[i][2] * smv[2]);
}

// ---------------- launch ----------------

extern "C" void kernel_launch(void* const* d_in, const int* in_sizes, int n_in,
                              void* d_out, int out_size, void* d_ws, size_t ws_size,
                              hipStream_t stream) {
    const float* source  = (const float*)d_in[0];  // (B,N,3)
    const float* target  = (const float*)d_in[1];  // (B,N,3)
    const float* fsource = (const float*)d_in[2];  // (B,F,N)
    const float* ftarget = (const float*)d_in[3];  // (B,F,N)
    float* out = (float*)d_out;                    // 16*9 Rot + 16*3 tr

    char* ws = (char*)d_ws;
    const size_t featBytes = (size_t)BB * NN * FF * 2;
    float*    gacc = (float*)ws;                                   // 16 x 16 f32
    uint16_t* Abf  = (uint16_t*)(ws + 4096);                       // (B,N,F) bf16
    uint16_t* Bbf  = (uint16_t*)(ws + 4096 + featBytes);           // (B,F,N) bf16
    float*    tgtT = (float*)(ws + 4096 + 2 * featBytes);          // (B,3,N) f32

    zero_kernel<<<1, 256, 0, stream>>>(gacc);
    {
        int n4 = BB * FF * NN / 4;
        cvt_kernel<<<(n4 + 255) / 256, 256, 0, stream>>>(fsource, Bbf, n4);
    }
    {
        dim3 g(NN / 32, FF / 32, BB), blk(32, 8);
        tr_kernel<<<g, blk, 0, stream>>>(ftarget, Abf);
    }
    tgt_kernel<<<(BB * NN + 255) / 256, 256, 0, stream>>>(target, tgtT);
    {
        dim3 g(NN / 128, BB);
        reg_kernel<<<g, 256, 0, stream>>>(Abf, Bbf, tgtT, source, gacc);
    }
    solve_kernel<<<1, 32, 0, stream>>>(gacc, out);
}
